// PointNet2Regressor_48447231098971
// MI455X (gfx1250) — compile-verified, matrix operand order checked by reading, not run on hardware
//
#include <hip/hip_runtime.h>
#include <hip/hip_bf16.h>

typedef __attribute__((ext_vector_type(16))) __bf16 v16bf;
typedef __attribute__((ext_vector_type(8)))  __bf16 v8bf;
typedef __attribute__((ext_vector_type(8)))  float  v8f;

#define BN_EPS 1e-5f

// =====================================================================
// WMMA GEMM: C[M,N] = A[M,Kp] * Bt[N,Kp]^T   (bf16 in, f32 out)
//   A : row-major bf16, K padded to Kp (mult of 32), zero-filled pad
//   Bt: W transposed+packed to bf16, shape N x Kp (row n = column n of W)
//   M % 64 == 0, N % 32 == 0 (guaranteed by this network's shapes)
// block = 128 threads = 4 waves; wave computes a 16x32 C strip
// (1 A fragment reused by 2 WMMAs). No LDS, no predication:
// per ISA layout, each lane's fragment = two contiguous 16B runs.
// =====================================================================
__global__ __launch_bounds__(128) void k_gemm_wmma(
    const __bf16* __restrict__ A, const __bf16* __restrict__ Bt,
    float* __restrict__ C, int M, int Kp, int N) {
  const int wave = threadIdx.x >> 5;
  const int lane = threadIdx.x & 31;
  const int hlf  = lane >> 4;       // half-wave select
  const int l15  = lane & 15;
  const int mRow  = blockIdx.x * 64 + wave * 16 + l15;  // this lane's A row
  const int nBase = blockIdx.y * 32;
  const int n0 = nBase + l15;
  const int n1 = n0 + 16;
  const __bf16* arow = A  + (size_t)mRow * Kp + hlf * 8;
  const __bf16* brow0 = Bt + (size_t)n0  * Kp + hlf * 8;
  const __bf16* brow1 = Bt + (size_t)n1  * Kp + hlf * 8;
  v8f acc0 = {}, acc1 = {};
  for (int k0 = 0; k0 < Kp; k0 += 32) {
    // speculative prefetch of next A K-slice -> global_prefetch_b8
    __builtin_prefetch(arow + k0 + 32, 0, 1);
    v8bf alo = *(const v8bf*)(arow + k0);        // K = k0+8h .. +7
    v8bf ahi = *(const v8bf*)(arow + k0 + 16);   // K = k0+16+8h .. +7
    v16bf af = __builtin_shufflevector(alo, ahi, 0, 1, 2, 3, 4, 5, 6, 7,
                                       8, 9, 10, 11, 12, 13, 14, 15);
    v8bf b0lo = *(const v8bf*)(brow0 + k0);
    v8bf b0hi = *(const v8bf*)(brow0 + k0 + 16);
    v16bf bf0 = __builtin_shufflevector(b0lo, b0hi, 0, 1, 2, 3, 4, 5, 6, 7,
                                        8, 9, 10, 11, 12, 13, 14, 15);
    v8bf b1lo = *(const v8bf*)(brow1 + k0);
    v8bf b1hi = *(const v8bf*)(brow1 + k0 + 16);
    v16bf bf1 = __builtin_shufflevector(b1lo, b1hi, 0, 1, 2, 3, 4, 5, 6, 7,
                                        8, 9, 10, 11, 12, 13, 14, 15);
    acc0 = __builtin_amdgcn_wmma_f32_16x16x32_bf16(
        false, af, false, bf0, (short)0, acc0, false, false);
    acc1 = __builtin_amdgcn_wmma_f32_16x16x32_bf16(
        false, af, false, bf1, (short)0, acc1, false, false);
  }
  // C layout: VGPR r -> row = tileBase + 8*hlf + r, col = lane&15
  const int rowBase = blockIdx.x * 64 + wave * 16 + hlf * 8;
#pragma unroll
  for (int r = 0; r < 8; ++r) {
    C[(size_t)(rowBase + r) * N + n0] = acc0[r];
    C[(size_t)(rowBase + r) * N + n1] = acc1[r];
  }
}

// ====== pack fp32 W[K,N] -> bf16 Wt[N,Kp] (transposed, zero-padded) ======
__global__ __launch_bounds__(256) void k_pack_w(
    const float* __restrict__ W, __bf16* __restrict__ Wt, int K, int N,
    int Kp) {
  int t = blockIdx.x * 256 + threadIdx.x;
  if (t >= N * Kp) return;
  int n = t / Kp, k = t % Kp;
  Wt[t] = (k < K) ? (__bf16)W[(size_t)k * N + n] : (__bf16)0.0f;
}

// ============ per-column mean/var over M rows (BatchNorm stats) ============
__global__ __launch_bounds__(256) void k_colstats(
    const float* __restrict__ X, float* __restrict__ mean,
    float* __restrict__ var, int M, int N) {
  const int n = blockIdx.x;
  float s = 0.0f, s2 = 0.0f;
  for (int m = threadIdx.x; m < M; m += 256) {
    float v = X[(size_t)m * N + n];
    s += v; s2 += v * v;
  }
  __shared__ float ls[256], ls2[256];
  ls[threadIdx.x] = s; ls2[threadIdx.x] = s2;
  __syncthreads();
  for (int off = 128; off > 0; off >>= 1) {
    if (threadIdx.x < off) {
      ls[threadIdx.x]  += ls[threadIdx.x + off];
      ls2[threadIdx.x] += ls2[threadIdx.x + off];
    }
    __syncthreads();
  }
  if (threadIdx.x == 0) {
    float mu = ls[0] / (float)M;
    mean[n] = mu;
    var[n]  = ls2[0] / (float)M - mu * mu;
  }
}

// ==== normalize+ReLU: fp32 in place AND bf16 (padded) copy for next GEMM ====
__global__ __launch_bounds__(256) void k_bn_relu(
    float* __restrict__ X, __bf16* __restrict__ Xb,
    const float* __restrict__ mean, const float* __restrict__ var,
    const float* __restrict__ gamma, const float* __restrict__ beta, int M,
    int N, int Np) {
  size_t i = (size_t)blockIdx.x * 256 + threadIdx.x;
  if (i >= (size_t)M * Np) return;
  int n = (int)(i % (size_t)Np);
  size_t m = i / (size_t)Np;
  if (n < N) {
    size_t xi = m * (size_t)N + n;
    float v = (X[xi] - mean[n]) * rsqrtf(var[n] + BN_EPS) * gamma[n] + beta[n];
    v = fmaxf(v, 0.0f);
    X[xi] = v;
    Xb[i] = (__bf16)v;
  } else {
    Xb[i] = (__bf16)0.0f;
  }
}

// ============ farthest point sampling: one block per batch ============
__global__ __launch_bounds__(256) void k_fps(
    const float* __restrict__ xyz, int ldx, int N, int npoint,
    int* __restrict__ idx_out) {
  const int b = blockIdx.x;
  const float* base = xyz + (size_t)b * N * ldx;
  __shared__ float dists[4096];
  __shared__ float redv[256];
  __shared__ int   redi[256];
  __shared__ int   cur;
  if (threadIdx.x == 0) cur = 0;
  for (int i = threadIdx.x; i < N; i += 256) dists[i] = 1e10f;
  __syncthreads();
  for (int it = 0; it < npoint; ++it) {
    int c = cur;
    if (threadIdx.x == 0) idx_out[b * npoint + it] = c;
    float cx = base[(size_t)c * ldx], cy = base[(size_t)c * ldx + 1],
          cz = base[(size_t)c * ldx + 2];
    float best = -1.0f; int bi = 0;
    for (int i = threadIdx.x; i < N; i += 256) {
      float dx = base[(size_t)i * ldx] - cx;
      float dy = base[(size_t)i * ldx + 1] - cy;
      float dz = base[(size_t)i * ldx + 2] - cz;
      float d = dx * dx + dy * dy + dz * dz;
      float nd = fminf(dists[i], d);
      dists[i] = nd;
      if (nd > best) { best = nd; bi = i; }
    }
    redv[threadIdx.x] = best; redi[threadIdx.x] = bi;
    __syncthreads();
    for (int off = 128; off > 0; off >>= 1) {
      if (threadIdx.x < off && redv[threadIdx.x + off] > redv[threadIdx.x]) {
        redv[threadIdx.x] = redv[threadIdx.x + off];
        redi[threadIdx.x] = redi[threadIdx.x + off];
      }
      __syncthreads();
    }
    if (threadIdx.x == 0) cur = redi[0];
    __syncthreads();
  }
}

// ============ gather sampled centroids ============
__global__ __launch_bounds__(256) void k_gather_xyz(
    const float* __restrict__ xyz, int ldx, const int* __restrict__ fidx,
    float* __restrict__ out, int total, int N, int npoint) {
  int t = blockIdx.x * 256 + threadIdx.x;
  if (t >= total) return;
  int b = t / npoint;
  int pi = fidx[t];
  const float* s = xyz + ((size_t)b * N + pi) * ldx;
  out[(size_t)t * 3]     = s[0];
  out[(size_t)t * 3 + 1] = s[1];
  out[(size_t)t * 3 + 2] = s[2];
}

// ============ ball query (first nsample in index order, pad w/ first) ======
__global__ __launch_bounds__(256) void k_ball_query(
    const float* __restrict__ xyz, int ldx, const float* __restrict__ new_xyz,
    int* __restrict__ idx_out, int total, int N, int npoint, float r2,
    int nsample) {
  int t = blockIdx.x * 256 + threadIdx.x;
  if (t >= total) return;
  int b = t / npoint;
  const float* base = xyz + (size_t)b * N * ldx;
  float qx = new_xyz[(size_t)t * 3], qy = new_xyz[(size_t)t * 3 + 1],
        qz = new_xyz[(size_t)t * 3 + 2];
  int* out = idx_out + (size_t)t * nsample;
  int cnt = 0, first = 0;
  for (int i = 0; i < N && cnt < nsample; ++i) {
    float dx = base[(size_t)i * ldx] - qx;
    float dy = base[(size_t)i * ldx + 1] - qy;
    float dz = base[(size_t)i * ldx + 2] - qz;
    float d = dx * dx + dy * dy + dz * dz;
    if (d <= r2) { if (cnt == 0) first = i; out[cnt++] = i; }
  }
  if (cnt == 0) { for (int s = 0; s < nsample; ++s) out[s] = 0; }
  else          { for (int s = cnt; s < nsample; ++s) out[s] = first; }
}

// ====== build grouped rows as padded bf16: [xyz - new_xyz, feats, 0...] =====
__global__ __launch_bounds__(256) void k_group(
    const float* __restrict__ xyz, int ldx, const float* __restrict__ feats,
    int ldf, int Cf, const float* __restrict__ new_xyz,
    const int* __restrict__ idx, __bf16* __restrict__ out, int Kp, int total,
    int N, int npoint, int nsample) {
  int t = blockIdx.x * 256 + threadIdx.x;
  if (t >= total) return;
  int bp = t / nsample;
  int b  = bp / npoint;
  int pi = idx[t];
  const float* src = xyz + ((size_t)b * N + pi) * ldx;
  __bf16* row = out + (size_t)t * Kp;
  row[0] = (__bf16)(src[0] - new_xyz[(size_t)bp * 3]);
  row[1] = (__bf16)(src[1] - new_xyz[(size_t)bp * 3 + 1]);
  row[2] = (__bf16)(src[2] - new_xyz[(size_t)bp * 3 + 2]);
  const float* fs = feats + ((size_t)b * N + pi) * ldf;
  for (int c = 0; c < Cf; ++c) row[3 + c] = (__bf16)fs[c];
  for (int c = 3 + Cf; c < Kp; ++c) row[c] = (__bf16)0.0f;
}

// ============ max over nsample axis ============
__global__ __launch_bounds__(256) void k_maxpool(
    const float* __restrict__ X, float* __restrict__ out, int total,
    int nsample, int C) {
  int t = blockIdx.x * 256 + threadIdx.x;
  if (t >= total) return;
  int c  = t % C;
  int bp = t / C;
  const float* base = X + (size_t)bp * nsample * C + c;
  float m = base[0];
  for (int s = 1; s < nsample; ++s) m = fmaxf(m, base[(size_t)s * C]);
  out[t] = m;
}

// ==== 3-NN inverse-distance interpolation + concat -> padded bf16 rows ====
__global__ __launch_bounds__(256) void k_fp_interp(
    const float* __restrict__ unk_xyz, int ldu,
    const float* __restrict__ kn_xyz, const float* __restrict__ kn_feats,
    int Ck, const float* __restrict__ unk_feats, int ldfu, int Cu,
    __bf16* __restrict__ out, int Kp, int total, int Nu, int Nk) {
  int t = blockIdx.x * 256 + threadIdx.x;
  if (t >= total) return;
  int b = t / Nu;
  const float* q = unk_xyz + (size_t)t * ldu;
  float qx = q[0], qy = q[1], qz = q[2];
  const float* kb = kn_xyz + (size_t)b * Nk * 3;
  float d0 = 1e30f, d1 = 1e30f, d2 = 1e30f;
  int   i0 = 0,     i1 = 0,     i2 = 0;
  for (int i = 0; i < Nk; ++i) {
    float dx = kb[(size_t)i * 3] - qx;
    float dy = kb[(size_t)i * 3 + 1] - qy;
    float dz = kb[(size_t)i * 3 + 2] - qz;
    float d = dx * dx + dy * dy + dz * dz;
    if (d < d0)      { d2 = d1; i2 = i1; d1 = d0; i1 = i0; d0 = d; i0 = i; }
    else if (d < d1) { d2 = d1; i2 = i1; d1 = d; i1 = i; }
    else if (d < d2) { d2 = d; i2 = i; }
  }
  float w0 = 1.0f / (d0 + 1e-8f);
  float w1 = 1.0f / (d1 + 1e-8f);
  float w2 = 1.0f / (d2 + 1e-8f);
  float ws = w0 + w1 + w2;
  w0 /= ws; w1 /= ws; w2 /= ws;
  const float* f0 = kn_feats + ((size_t)b * Nk + i0) * Ck;
  const float* f1 = kn_feats + ((size_t)b * Nk + i1) * Ck;
  const float* f2 = kn_feats + ((size_t)b * Nk + i2) * Ck;
  __bf16* row = out + (size_t)t * Kp;
  for (int c = 0; c < Ck; ++c)
    row[c] = (__bf16)(w0 * f0[c] + w1 * f1[c] + w2 * f2[c]);
  const float* uf = unk_feats + (size_t)t * ldfu;
  for (int c = 0; c < Cu; ++c) row[Ck + c] = (__bf16)uf[c];
  for (int c = Ck + Cu; c < Kp; ++c) row[c] = (__bf16)0.0f;
}

// =====================================================================
// Host orchestration
// =====================================================================
extern "C" void kernel_launch(void* const* d_in, const int* in_sizes, int n_in,
                              void* d_out, int out_size, void* d_ws,
                              size_t ws_size, hipStream_t stream) {
  (void)in_sizes; (void)n_in; (void)out_size; (void)ws_size;
  const float* pc = (const float*)d_in[0];   // (32, 4096, 6)

  struct Layer { const float *W, *g, *b; };
  Layer saL[4][3], fpL[4][3];
  const int fpN[4] = {3, 2, 2, 2};
  int p = 1;
  for (int i = 0; i < 4; ++i)
    for (int l = 0; l < 3; ++l) {
      saL[i][l] = { (const float*)d_in[p], (const float*)d_in[p + 1],
                    (const float*)d_in[p + 2] };
      p += 3;
    }
  for (int i = 0; i < 4; ++i)
    for (int l = 0; l < fpN[i]; ++l) {
      fpL[i][l] = { (const float*)d_in[p], (const float*)d_in[p + 1],
                    (const float*)d_in[p + 2] };
      p += 3;
    }

  const int Bn = 32;
  const int Npts[5]    = {4096, 256, 128, 64, 16};
  const float radii[4] = {0.2f, 0.4f, 0.4f, 0.8f};
  const int NS = 32;
  const int saDims[4][4] = {{6, 32, 64, 128}, {131, 128, 128, 256},
                            {259, 256, 256, 512}, {515, 512, 512, 1024}};
  const int fpDims[4][4] = {{259, 256, 256, 256}, {640, 512, 256, 0},
                            {768, 512, 512, 0},   {1536, 512, 512, 0}};
  auto pad32 = [](int k) { return (k + 31) & ~31; };

  // ---- workspace bump allocator (needs ~310 MB) ----
  char* wp = (char*)d_ws;
  auto alloc = [&](size_t bytes) {
    void* q = wp;
    wp += (bytes + 255) & ~(size_t)255;
    return q;
  };
  const size_t C_ELEMS  = 33554432;  // 262144*128 == 131072*256 (max fp32 act)
  const size_t BB_ELEMS = 37748736;  // 131072*288 (max padded bf16 act)
  float*  Cbuf = (float*)alloc(C_ELEMS * sizeof(float));
  __bf16* bb0  = (__bf16*)alloc(BB_ELEMS * sizeof(__bf16));
  __bf16* bb1  = (__bf16*)alloc(BB_ELEMS * sizeof(__bf16));
  __bf16* wPack = (__bf16*)alloc(786432 * sizeof(__bf16));  // max N*Kp
  float* xyzL[5];  xyzL[0] = nullptr;
  float* featsL[5]; featsL[0] = nullptr;
  const size_t xyzSz[5]   = {0, 32u*256*3, 32u*128*3, 32u*64*3, 32u*16*3};
  const size_t featsSz[5] = {0, 32u*256*256, 32u*128*512, 32u*64*512, 32u*16*1024};
  for (int i = 1; i < 5; ++i) {
    xyzL[i]   = (float*)alloc(xyzSz[i] * sizeof(float));
    featsL[i] = (float*)alloc(featsSz[i] * sizeof(float));
  }
  float* mean = (float*)alloc(1024 * sizeof(float));
  float* var  = (float*)alloc(1024 * sizeof(float));
  int* fidx   = (int*)alloc(32u * 256 * sizeof(int));
  int* nidx   = (int*)alloc(32u * 256 * 32 * sizeof(int));

  // Runs an MLP whose padded-bf16 input rows are in `Ain`. Final layer fp32
  // activation is written+normalized in place in `finalOut` (or Cbuf).
  // Returns pointer to the final fp32 activation matrix.
  auto run_mlp = [&](__bf16* Ain, int rows, const Layer* Ls, const int* dims,
                     int nlayers, float* finalOut) -> float* {
    __bf16* cur = Ain;
    float* Cdst = Cbuf;
    for (int l = 0; l < nlayers; ++l) {
      int K = dims[l], Nc = dims[l + 1], Kp = pad32(K);
      k_pack_w<<<(Nc * Kp + 255) / 256, 256, 0, stream>>>(Ls[l].W, wPack, K,
                                                          Nc, Kp);
      Cdst = (l == nlayers - 1 && finalOut) ? finalOut : Cbuf;
      dim3 g((unsigned)(rows / 64), (unsigned)(Nc / 32));
      k_gemm_wmma<<<g, 128, 0, stream>>>(cur, wPack, Cdst, rows, Kp, Nc);
      k_colstats<<<Nc, 256, 0, stream>>>(Cdst, mean, var, rows, Nc);
      __bf16* nxt = (cur == bb0) ? bb1 : bb0;
      size_t tot = (size_t)rows * Nc;  // Nc already multiple of 32
      k_bn_relu<<<(unsigned)((tot + 255) / 256), 256, 0, stream>>>(
          Cdst, nxt, mean, var, Ls[l].g, Ls[l].b, rows, Nc, Nc);
      cur = nxt;
    }
    return Cdst;
  };

  // ============================ SA stages ============================
  int featC = 3;  // channels of l_feats[i]
  for (int i = 0; i < 4; ++i) {
    const int Nin = Npts[i], Nout = Npts[i + 1];
    const float* xyz_in = (i == 0) ? pc : xyzL[i];
    const int ldx = (i == 0) ? 6 : 3;
    const float* feats_in = (i == 0) ? (pc + 3) : featsL[i];
    const int ldf = (i == 0) ? 6 : featC;

    k_fps<<<Bn, 256, 0, stream>>>(xyz_in, ldx, Nin, Nout, fidx);
    int tc = Bn * Nout;
    k_gather_xyz<<<(tc + 255) / 256, 256, 0, stream>>>(
        xyz_in, ldx, fidx, xyzL[i + 1], tc, Nin, Nout);
    k_ball_query<<<(tc + 255) / 256, 256, 0, stream>>>(
        xyz_in, ldx, xyzL[i + 1], nidx, tc, Nin, Nout, radii[i] * radii[i],
        NS);
    int rows = Bn * Nout * NS;
    int Kp0 = pad32(saDims[i][0]);
    k_group<<<(rows + 255) / 256, 256, 0, stream>>>(
        xyz_in, ldx, feats_in, ldf, featC, xyzL[i + 1], nidx, bb0, Kp0, rows,
        Nin, Nout, NS);

    float* y = run_mlp(bb0, rows, saL[i], saDims[i], 3, nullptr);
    int Cout = saDims[i][3];
    int tp = Bn * Nout * Cout;
    k_maxpool<<<(tp + 255) / 256, 256, 0, stream>>>(y, featsL[i + 1], tp, NS,
                                                    Cout);
    featC = Cout;
  }

  // ============================ FP stages ============================
  int lvlC[5] = {3, 128, 256, 512, 1024};
  for (int j = 0; j < 4; ++j) {
    int fpIdx  = 3 - j;       // params['fp'][-1-j]
    int unkLvl = 3 - j;       // l_feats[unkLvl] gets replaced
    int knLvl  = unkLvl + 1;
    int Nu = Npts[unkLvl], Nk = Npts[knLvl];
    const float* unk_xyz = (unkLvl == 0) ? pc : xyzL[unkLvl];
    int ldu = (unkLvl == 0) ? 6 : 3;
    const float* unk_feats = (unkLvl == 0) ? (pc + 3) : featsL[unkLvl];
    int Cu = lvlC[unkLvl];
    int ldfu = (unkLvl == 0) ? 6 : Cu;
    const float* kn_feats = featsL[knLvl];
    int Ck = lvlC[knLvl];

    int rows = Bn * Nu;
    int Kp0 = pad32(fpDims[fpIdx][0]);
    k_fp_interp<<<(rows + 255) / 256, 256, 0, stream>>>(
        unk_xyz, ldu, xyzL[knLvl], kn_feats, Ck, unk_feats, ldfu, Cu, bb0,
        Kp0, rows, Nu, Nk);

    float* dst = (unkLvl == 0) ? (float*)d_out : featsL[unkLvl];
    run_mlp(bb0, rows, fpL[fpIdx], fpDims[fpIdx], fpN[fpIdx], dst);
    lvlC[unkLvl] = fpDims[fpIdx][fpN[fpIdx]];
  }
}
